// GraphAttentionLayer_266287972501
// MI455X (gfx1250) — compile-verified, matrix operand order checked by reading, not run on hardware
//
#include <hip/hip_runtime.h>
#include <hip/hip_bf16.h>

typedef __attribute__((ext_vector_type(16))) _Float16 v16h;
typedef __attribute__((ext_vector_type(8)))  _Float16 v8h;
typedef __attribute__((ext_vector_type(8)))  float    v8f;
typedef __attribute__((ext_vector_type(4)))  float    v4f;
typedef __attribute__((ext_vector_type(4)))  int      v4i;

#define GAT_N     8192
#define GAT_DIN   512
#define GAT_DOUT  64
#define NEG_INF_F (-1e15f)

// ---------------------------------------------------------------------------
// Kernel 1: h = X @ W  (f32 in, f32 h out + transposed f16 copy hT[64][8192])
// One wave per 16-row x 64-col tile, v_wmma_f32_16x16x32_f16 K-loop.
// ---------------------------------------------------------------------------
__global__ void k_h(const float* __restrict__ X, const float* __restrict__ W,
                    float* __restrict__ h, _Float16* __restrict__ hT)
{
    const int wave = threadIdx.x >> 5;
    const int lane = threadIdx.x & 31;
    const int half = lane >> 4;
    const int l16  = lane & 15;
    const int i0   = (blockIdx.x * 4 + wave) * 16;

    v8f acc[4] = {};

    for (int k0 = 0; k0 < GAT_DIN; k0 += 32) {
        const float* arow = X + (size_t)(i0 + l16) * GAT_DIN + k0 + 16 * half;
        v16h a;
        #pragma unroll
        for (int t = 0; t < 16; ++t) a[t] = (_Float16)arow[t];

        #pragma unroll
        for (int nb = 0; nb < 4; ++nb) {
            const float* bcol = W + (size_t)(k0 + 16 * half) * GAT_DOUT + nb * 16 + l16;
            v16h b;
            #pragma unroll
            for (int t = 0; t < 16; ++t) b[t] = (_Float16)bcol[(size_t)t * GAT_DOUT];
            acc[nb] = __builtin_amdgcn_wmma_f32_16x16x32_f16(
                false, a, false, b, (short)0, acc[nb], false, false);
        }
    }

    #pragma unroll
    for (int nb = 0; nb < 4; ++nb) {
        const int col = nb * 16 + l16;
        v8h hv;
        #pragma unroll
        for (int r = 0; r < 8; ++r) {
            const int row = i0 + r + 8 * half;
            h[(size_t)row * GAT_DOUT + col] = acc[nb][r];
            hv[r] = (_Float16)acc[nb][r];
        }
        *(v8h*)(hT + (size_t)col * GAT_N + i0 + 8 * half) = hv;
    }
}

// ---------------------------------------------------------------------------
// Kernel 2: s_src[i] = h[i,:].a_src ; s_dst[i] = h[i,:].a_dst (wave32 reduce)
// ---------------------------------------------------------------------------
__global__ void k_scores(const float* __restrict__ h,
                         const float* __restrict__ a_src,
                         const float* __restrict__ a_dst,
                         float* __restrict__ s_src, float* __restrict__ s_dst)
{
    const int gwave = (int)((blockIdx.x * blockDim.x + threadIdx.x) >> 5);
    const int lane  = threadIdx.x & 31;
    if (gwave >= GAT_N) return;

    const float* row = h + (size_t)gwave * GAT_DOUT;
    float ps = 0.f, pd = 0.f;
    #pragma unroll
    for (int t = 0; t < 2; ++t) {
        const int c = lane + 32 * t;
        const float v = row[c];
        ps += v * a_src[c];
        pd += v * a_dst[c];
    }
    #pragma unroll
    for (int off = 16; off > 0; off >>= 1) {
        ps += __shfl_xor(ps, off, 32);
        pd += __shfl_xor(pd, off, 32);
    }
    if (lane == 0) { s_src[gwave] = ps; s_dst[gwave] = pd; }
}

// ---------------------------------------------------------------------------
// Kernel 3: fused masked-softmax attention, split-j flash style.
// 1 workgroup (4 waves) per 16-row block; each wave owns a 2048-column slice.
// Adjacency is double-buffered into LDS with GLOBAL_LOAD_ASYNC_TO_LDS_B128
// (ASYNCcnt path); partial (m, l, acc) merged across waves via LDS +
// ds_add_f32 atomics.
// ---------------------------------------------------------------------------
#define JSLICE 2048

__global__ __launch_bounds__(128) void k_attn(
    const int* __restrict__ adj, const _Float16* __restrict__ hT,
    const float* __restrict__ s_src, const float* __restrict__ s_dst,
    float* __restrict__ out)
{
    __shared__ int   adjbuf[4][2][16 * 32];   // per-wave double buffer (16KB)
    __shared__ float sm[4][16];               // per-wave row max
    __shared__ float sl[4][16];               // per-wave row sum
    __shared__ float stile[16][64];           // merged accumulator (4KB)

    const int wave = threadIdx.x >> 5;
    const int lane = threadIdx.x & 31;
    const int half = lane >> 4;
    const int l16  = lane & 15;
    const int i0   = blockIdx.x * 16;

    const int   myrow  = i0 + l16;
    const float ssrc   = s_src[myrow];
    const int*  adjrow = adj + (size_t)myrow * GAT_N;
    const int   jstart = wave * JSLICE;

    // zero the merge tile before the first barrier
    #pragma unroll
    for (int t = 0; t < 8; ++t)
        ((float*)stile)[threadIdx.x * 8 + t] = 0.f;

    // per-lane LDS destinations for the two buffers (64B strip per lane)
    const unsigned lds0 = (unsigned)(uintptr_t)&adjbuf[wave][0][lane * 16];
    const unsigned lds1 = (unsigned)(uintptr_t)&adjbuf[wave][1][lane * 16];

    // issue one 64B async strip: 4 x global_load_async_to_lds_b128
    #define ISSUE_STRIP(LDSA, GPTR)                                            \
        do {                                                                   \
            unsigned long long _ga = (unsigned long long)(uintptr_t)(GPTR);    \
            unsigned _la = (LDSA);                                             \
            asm volatile("global_load_async_to_lds_b128 %0, %1, off"           \
                         :: "v"(_la), "v"(_ga) : "memory");                    \
            asm volatile("global_load_async_to_lds_b128 %0, %1, off offset:16" \
                         :: "v"(_la), "v"(_ga) : "memory");                    \
            asm volatile("global_load_async_to_lds_b128 %0, %1, off offset:32" \
                         :: "v"(_la), "v"(_ga) : "memory");                    \
            asm volatile("global_load_async_to_lds_b128 %0, %1, off offset:48" \
                         :: "v"(_la), "v"(_ga) : "memory");                    \
        } while (0)

    float m = -INFINITY;
    float l = 0.f;
    v8f acc[4] = {};

    // prologue: start chunk 0 of this wave's slice
    ISSUE_STRIP(lds0, adjrow + jstart + 16 * half);

    int buf = 0;
    for (int j0 = jstart; j0 < jstart + JSLICE; j0 += 32) {
        const int jbase = j0 + 16 * half;

        // issue next chunk into the other buffer (tail clamps to a valid addr)
        const int jn = (j0 + 32 < jstart + JSLICE) ? (j0 + 32) : jstart;
        ISSUE_STRIP(buf ? lds0 : lds1, adjrow + jn + 16 * half);

        // wait for current chunk's 4 strips (4 newer ones may stay in flight)
        asm volatile("s_wait_asynccnt 0x4" ::: "memory");

        __builtin_prefetch(s_dst + jbase + 32, 0, 0);  // global_prefetch_b8

        // --- masked leaky-relu logits from the LDS-staged adjacency ---
        const v4i* ap = (const v4i*)&adjbuf[wave][buf][lane * 16]; // ds_load_b128
        const v4f* sp = (const v4f*)(s_dst + jbase);
        float e[16];
        #pragma unroll
        for (int q = 0; q < 4; ++q) {
            const v4i a4 = ap[q];
            const v4f s4 = sp[q];
            #pragma unroll
            for (int t = 0; t < 4; ++t) {
                float x = ssrc + s4[t];
                x = x > 0.f ? x : 0.2f * x;           // leaky_relu(0.2)
                e[q * 4 + t] = (a4[t] > 0) ? x : NEG_INF_F;
            }
        }

        // --- online softmax (lane-local + partner-lane via shfl_xor 16) ---
        float cmax = e[0];
        #pragma unroll
        for (int t = 1; t < 16; ++t) cmax = fmaxf(cmax, e[t]);
        cmax = fmaxf(cmax, __shfl_xor(cmax, 16, 32));
        const float mnew = fmaxf(m, cmax);
        const float corr = __expf(m - mnew);

        v16h p;
        float psum = 0.f;
        #pragma unroll
        for (int t = 0; t < 16; ++t) {
            const float pv = __expf(e[t] - mnew);
            psum += pv;
            p[t] = (_Float16)pv;
        }
        psum += __shfl_xor(psum, 16, 32);
        l = l * corr + psum;
        m = mnew;

        // rescale accumulators (row of acc VGPR r is M = r + 8*half)
        #pragma unroll
        for (int r = 0; r < 8; ++r) {
            const float cr = __shfl(corr, r + 8 * half, 32);
            #pragma unroll
            for (int nb = 0; nb < 4; ++nb) acc[nb][r] *= cr;
        }

        // acc += P(16x32) @ hT-chunk(32x64): 4 WMMAs
        #pragma unroll
        for (int nb = 0; nb < 4; ++nb) {
            const v16h b = *(const v16h*)(hT + (size_t)(nb * 16 + l16) * GAT_N + jbase);
            acc[nb] = __builtin_amdgcn_wmma_f32_16x16x32_f16(
                false, p, false, b, (short)0, acc[nb], false, false);
        }

        buf ^= 1;
    }
    asm volatile("s_wait_asynccnt 0x0" ::: "memory");  // drain tail strips
    #undef ISSUE_STRIP

    // --- cross-wave merge of (m, l, acc) ---
    if (half == 0) { sm[wave][l16] = m; sl[wave][l16] = l; }
    __syncthreads();

    // global max for this lane's row, and this wave's rescale factor
    float M = -INFINITY;
    #pragma unroll
    for (int w = 0; w < 4; ++w) M = fmaxf(M, sm[w][l16]);
    const float f = __expf(m - M);

    #pragma unroll
    for (int r = 0; r < 8; ++r) {
        const float fr = __shfl(f, r + 8 * half, 32);
        #pragma unroll
        for (int nb = 0; nb < 4; ++nb)
            atomicAdd(&stile[r + 8 * half][nb * 16 + l16], acc[nb][r] * fr); // ds_add_f32
    }
    __syncthreads();

    // --- epilogue: each thread finishes 8 contiguous outputs of one row ---
    const int row  = threadIdx.x >> 3;          // 0..15
    const int col0 = (threadIdx.x & 7) * 8;
    float Mr = -INFINITY;
    #pragma unroll
    for (int w = 0; w < 4; ++w) Mr = fmaxf(Mr, sm[w][row]);
    float Lr = 0.f;
    #pragma unroll
    for (int w = 0; w < 4; ++w) Lr += __expf(sm[w][row] - Mr) * sl[w][row];
    const float linv = 1.f / Lr;

    #pragma unroll
    for (int c = 0; c < 8; ++c) {
        float v = stile[row][col0 + c] * linv;
        v = v > 0.f ? v : (__expf(v) - 1.f);    // ELU(alpha=1)
        out[(size_t)(i0 + row) * GAT_DOUT + col0 + c] = v;
    }
}

// ---------------------------------------------------------------------------
extern "C" void kernel_launch(void* const* d_in, const int* in_sizes, int n_in,
                              void* d_out, int out_size, void* d_ws, size_t ws_size,
                              hipStream_t stream)
{
    (void)in_sizes; (void)n_in; (void)out_size; (void)ws_size;

    const float* X     = (const float*)d_in[0];   // [8192,512]
    const int*   adj   = (const int*)  d_in[1];   // [8192,8192]
    const float* W     = (const float*)d_in[2];   // [512,64]
    const float* a_src = (const float*)d_in[3];   // [64]
    const float* a_dst = (const float*)d_in[4];   // [64]
    float*       out   = (float*)d_out;           // [8192,64]

    float*     h     = (float*)d_ws;                        // 8192*64 f32
    _Float16*  hT    = (_Float16*)(h + GAT_N * GAT_DOUT);   // 64*8192 f16
    float*     s_src = (float*)(hT + (size_t)GAT_DOUT * GAT_N);
    float*     s_dst = s_src + GAT_N;

    k_h<<<dim3(GAT_N / 16 / 4), dim3(128), 0, stream>>>(X, W, h, hT);
    k_scores<<<dim3(GAT_N * 32 / 256), dim3(256), 0, stream>>>(h, a_src, a_dst, s_src, s_dst);
    // one workgroup (4 waves, split-j) per 16-row block
    k_attn<<<dim3(GAT_N / 16), dim3(128), 0, stream>>>(adj, hT, s_src, s_dst, out);
}
